// LinearAttention_27487790694454
// MI455X (gfx1250) — matrix operand order repaired and checked by reading, not compiled, not run on hardware
//
#include <hip/hip_runtime.h>
#include <hip/hip_bf16.h>

// ---------------- problem constants (from reference) ----------------
#define BB   2
#define FF   768
#define II   1536
#define EE   8
#define KC   5
#define GG   8
#define SS   1024
#define O0   (3 * II)          // 4608
#define NT   (BB * SS)         // 2048 tokens
#define ALPHA 0.02f

typedef __attribute__((ext_vector_type(16))) __bf16 v16bf;
typedef __attribute__((ext_vector_type(8)))  float  v8f;
typedef __attribute__((ext_vector_type(4)))  float  f4;

// =====================================================================
// 0) x [B,F,S] -> xt [token][F]  (32x32 LDS tile transpose, coalesced both sides)
// =====================================================================
__global__ __launch_bounds__(256) void transpose_xt_kernel(
    const float* __restrict__ x, float* __restrict__ xt)
{
    __shared__ float tile[32][33];
    int bid = blockIdx.x;
    const int st = bid & (SS / 32 - 1); bid >>= 5;        // 32 s-tiles
    const int ct = bid % (FF / 32);
    const int b  = bid / (FF / 32);
    const int s0 = st * 32, c0 = ct * 32;
    const int tx = threadIdx.x & 31, ty = threadIdx.x >> 5;   // ty: 0..7
#pragma unroll
    for (int r = 0; r < 32; r += 8)
        tile[ty + r][tx] = x[((size_t)b * FF + c0 + ty + r) * SS + s0 + tx];
    __syncthreads();
#pragma unroll
    for (int r = 0; r < 32; r += 8)
        xt[((size_t)b * SS + s0 + ty + r) * FF + c0 + tx] = tile[tx][ty + r];
}

// =====================================================================
// 1) top-1 gate + softmax + expert compaction. One wave per token.
//    xin is TOKEN-MAJOR [token][C] -> per-lane reads are coalesced.
// =====================================================================
__global__ __launch_bounds__(256) void gate_route_kernel(
    const float* __restrict__ xin,   // [NT, C]
    const float* __restrict__ wg,    // [C, E]
    int C,
    float* __restrict__ gates,       // [E, NT]
    int* __restrict__ cnt,           // [E]
    int* __restrict__ list)          // [E, NT]
{
    const int wid  = threadIdx.x >> 5;
    const int lane = threadIdx.x & 31;
    const int t = blockIdx.x * 8 + wid;
    if (t >= NT) return;

    float acc[EE];
#pragma unroll
    for (int e = 0; e < EE; ++e) acc[e] = 0.f;

    for (int c = lane; c < C; c += 32) {
        const float xv = xin[(size_t)t * C + c];
#pragma unroll
        for (int e = 0; e < EE; ++e) acc[e] += xv * wg[c * EE + e];
    }
#pragma unroll
    for (int off = 16; off > 0; off >>= 1) {
#pragma unroll
        for (int e = 0; e < EE; ++e) acc[e] += __shfl_xor(acc[e], off, 32);
    }
    float mx = acc[0]; int am = 0;
#pragma unroll
    for (int e = 1; e < EE; ++e) if (acc[e] > mx) { mx = acc[e]; am = e; }
    float ex[EE]; float den = 0.f;
#pragma unroll
    for (int e = 0; e < EE; ++e) { ex[e] = __expf(acc[e] - mx); den += ex[e]; }
    if (lane < EE) gates[lane * NT + t] = ex[lane] / den;
    if (lane == 0) {
        const int pos = atomicAdd(&cnt[am], 1);
        list[am * NT + pos] = t;
    }
}

// =====================================================================
// 2) grouped-GEMM for MoE experts, direct-to-fragment (no LDS staging).
//    D[o,tok] = sum_c W[e][c][o] * XT[tok][c]
//    block = 8 waves, tile M=128 x N=64 tokens, K-step 32, bf16 WMMA.
//    Ragged token slots are clamped to a valid row (finite data) and the
//    bogus columns are simply not stored -> no mask math in the hot loop.
// =====================================================================
__global__ __launch_bounds__(256) void moe_gemm_kernel(
    const float* __restrict__ w,     // [E, C, O]
    const float* __restrict__ xt,    // [NT, C] token-major
    const int* __restrict__ list,    // [E, NT]
    const int* __restrict__ cnt,     // [E]
    float* __restrict__ out,         // [B, O, S]
    int C, int O, int mtiles, int ntiles)
{
    int bid = blockIdx.x;
    const int e  = bid / (mtiles * ntiles);
    const int rm = bid % (mtiles * ntiles);
    const int mt = rm / ntiles;
    const int nt = rm % ntiles;
    const int cntE = cnt[e];
    if (nt * 64 >= cntE) return;

    const int tid  = threadIdx.x;
    const int wid  = tid >> 5;
    const int lane = tid & 31;
    const int lm   = lane & 15;
    const int hi   = lane >> 4;

    // per-lane token rows for the 4 N-subtiles (branchless ragged handling)
    int trow[4]; bool tvalid[4];
#pragma unroll
    for (int nsub = 0; nsub < 4; ++nsub) {
        const int ng  = nt * 64 + nsub * 16 + lm;
        const bool v  = ng < cntE;
        const int idx = v ? ng : (cntE - 1);       // cntE >= 1 guaranteed
        trow[nsub]   = list[e * NT + idx];
        tvalid[nsub] = v;
    }

    v8f acc[4];
#pragma unroll
    for (int n = 0; n < 4; ++n)
#pragma unroll
        for (int i = 0; i < 8; ++i) acc[n][i] = 0.f;

    const int m     = mt * 128 + wid * 16 + lm;
    const int abase = hi * 16;                     // A-frag K base (contiguous layout)
    const int kb    = hi * 16;                     // B-frag K base
    const float* wcol = w + (size_t)e * C * O + m;

    for (int c0 = 0; c0 < C; c0 += 32) {
        if (c0 + 32 < C)
            __builtin_prefetch(wcol + (size_t)(c0 + 32 + abase) * O, 0, 1);
        // ---- A fragment: 16x32 bf16; element i <-> K = 16*hi + i ----
        v16bf a;
#pragma unroll
        for (int i = 0; i < 16; ++i) {
            const int k = abase + i;
            a[i] = (__bf16)wcol[(size_t)(c0 + k) * O];
        }
        // ---- 4 WMMAs, B fragments straight from token-major rows ----
#pragma unroll
        for (int nsub = 0; nsub < 4; ++nsub) {
            const f4* bp = (const f4*)(xt + (size_t)trow[nsub] * C + c0 + kb);
            v16bf bf;
#pragma unroll
            for (int j = 0; j < 4; ++j) {
                const f4 bv = bp[j];
#pragma unroll
                for (int l = 0; l < 4; ++l) bf[j * 4 + l] = (__bf16)bv[l];
            }
            acc[nsub] = __builtin_amdgcn_wmma_f32_16x16x32_bf16(
                false, a, false, bf, (short)0, acc[nsub], false, false);
        }
    }

    // ---- masked scatter-store (C/D layout: vgpr r -> M=r (+8 for hi lanes)) ----
#pragma unroll
    for (int nsub = 0; nsub < 4; ++nsub) {
        if (!tvalid[nsub]) continue;
        const int t = trow[nsub];
        const int b = t / SS, s = t % SS;
#pragma unroll
        for (int r = 0; r < 8; ++r) {
            const int mm = mt * 128 + wid * 16 + r + hi * 8;
            out[((size_t)b * O + mm) * SS + s] = acc[nsub][r];
        }
    }
}

// =====================================================================
// 3) causal grouped conv as per-(b,group) GEMM: M=576, K=5*192=960.
//    Here the causal zero-pad DOES feed valid outputs -> keep the mask.
// =====================================================================
__global__ __launch_bounds__(256) void conv_gemm_kernel(
    const float* __restrict__ w1,    // [K, I/G, 3I]
    const float* __restrict__ g1t,   // [NT, I] token-major
    float* __restrict__ out)         // [B, 3I, S] channel-major
{
    constexpr int OG = O0 / GG;      // 576
    constexpr int CG = II / GG;      // 192  (multiple of 32 -> tap uniform per chunk)

    int bid = blockIdx.x;
    const int nt = bid & 15;  bid >>= 4;     // 16 token tiles (S/64)
    const int mt = bid % 5;   bid /= 5;      // 5 M tiles (ceil(576/128))
    const int g  = bid % GG;
    const int b  = bid / GG;

    const int tid  = threadIdx.x;
    const int wid  = tid >> 5;
    const int lane = tid & 31;
    const int lm   = lane & 15;
    const int hi   = lane >> 4;
    const int s0   = nt * 64;

    v8f acc[4];
#pragma unroll
    for (int n = 0; n < 4; ++n)
#pragma unroll
        for (int i = 0; i < 8; ++i) acc[n][i] = 0.f;

    const int ml = mt * 128 + wid * 16 + lm;
    const bool mvalid = ml < OG;             // wave-uniform (576 % 16 == 0)
    const int abase = hi * 16, kb = hi * 16;
    const float* wcol = w1 + (size_t)(g * OG + ml);

    for (int c0 = 0; c0 < KC * CG; c0 += 32) {     // 30 k-steps
        const int tap = c0 / CG;                   // uniform within chunk
        const int cib = c0 % CG;
        v16bf a;
#pragma unroll
        for (int i = 0; i < 16; ++i) {
            const int k = abase + i;
            const float v = mvalid ? wcol[(size_t)(tap * CG + cib + k) * O0] : 0.f;
            a[i] = (__bf16)v;
        }
#pragma unroll
        for (int nsub = 0; nsub < 4; ++nsub) {
            const int n  = nsub * 16 + lm;
            const int sp = s0 + n - 4 + tap;       // causal shift
            const float mk = sp >= 0 ? 1.f : 0.f;
            const int  spc = sp >= 0 ? sp : 0;
            const f4* bp = (const f4*)(g1t + ((size_t)b * SS + spc) * II
                                       + g * CG + cib + kb);
            v16bf bf;
#pragma unroll
            for (int j = 0; j < 4; ++j) {
                const f4 bv = bp[j] * mk;
#pragma unroll
                for (int l = 0; l < 4; ++l) bf[j * 4 + l] = (__bf16)bv[l];
            }
            acc[nsub] = __builtin_amdgcn_wmma_f32_16x16x32_bf16(
                false, a, false, bf, (short)0, acc[nsub], false, false);
        }
    }

#pragma unroll
    for (int nsub = 0; nsub < 4; ++nsub) {
        const int s = s0 + nsub * 16 + lm;
#pragma unroll
        for (int r = 0; r < 8; ++r) {
            const int mlr = mt * 128 + wid * 16 + r + hi * 8;
            if (mlr < OG)
                out[((size_t)b * O0 + g * OG + mlr) * SS + s] = acc[nsub][r];
        }
    }
}

// =====================================================================
// 4) cumsum along S + divide: one wave per (b,i) row, shuffle scan.
// =====================================================================
__global__ __launch_bounds__(256) void cumsum_div_kernel(
    const float* __restrict__ h0,    // [B, 3I, S] (depth|scale|shift)
    float* __restrict__ y)           // [B, I, S]
{
    const int wid  = threadIdx.x >> 5;
    const int lane = threadIdx.x & 31;
    const int row  = blockIdx.x * 8 + wid;
    if (row >= BB * II) return;
    const int b = row / II, i = row % II;

    const float* dp = h0 + ((size_t)b * O0 + i) * SS;
    const float* sc = h0 + ((size_t)b * O0 + II + i) * SS;
    const float* sh = h0 + ((size_t)b * O0 + 2 * II + i) * SS;
    float* yp = y + ((size_t)b * II + i) * SS;

    float carry = 0.f;
    for (int sb = 0; sb < SS; sb += 32) {
        const int s = sb + lane;
        float scan = dp[s];
#pragma unroll
        for (int off = 1; off < 32; off <<= 1) {
            const float nv = __shfl_up(scan, off, 32);
            if (lane >= off) scan += nv;
        }
        const float cum = carry + scan;
        yp[s] = cum / ((float)(s + 1) * sc[s] + sh[s]);
        carry += __shfl(scan, 31, 32);
    }
}

// =====================================================================
// 5) RMS-norm over channels + leaky; channel-major in, TOKEN-MAJOR out.
//    Block = (b, 32-token slab). Coalesced reads; 32x32 transposed writes.
// =====================================================================
__global__ __launch_bounds__(256) void rms_leaky_t_kernel(
    const float* __restrict__ y,     // [B, I, S]
    float* __restrict__ g1t)         // [NT, I]
{
    __shared__ float red[8][33];
    __shared__ float rinv[32];
    __shared__ float tile[32][33];
    const int b  = blockIdx.x / (SS / 32);
    const int s0 = (blockIdx.x % (SS / 32)) * 32;
    const int tx = threadIdx.x & 31;   // token within slab
    const int cg = threadIdx.x >> 5;   // 0..7 channel group

    float ss = 0.f;
    for (int i = cg; i < II; i += 8) {
        const float v = y[((size_t)b * II + i) * SS + s0 + tx];
        ss += v * v;
    }
    red[cg][tx] = ss; __syncthreads();
    if (threadIdx.x < 32) {
        float tot = 0.f;
#pragma unroll
        for (int g = 0; g < 8; ++g) tot += red[g][threadIdx.x];
        rinv[threadIdx.x] = rsqrtf(tot / (float)II + 1e-6f);
    }
    __syncthreads();

    for (int i0 = 0; i0 < II; i0 += 32) {
#pragma unroll
        for (int r = 0; r < 32; r += 8) {
            const float v = y[((size_t)b * II + i0 + cg + r) * SS + s0 + tx] * rinv[tx];
            tile[cg + r][tx] = v > 0.f ? v : ALPHA * v;
        }
        __syncthreads();
#pragma unroll
        for (int r = 0; r < 32; r += 8)
            g1t[((size_t)b * SS + s0 + cg + r) * II + i0 + tx] = tile[tx][cg + r];
        __syncthreads();
    }
}

// 5b) split_norm: z = s0*s1+sh over channel triplets, rms + leaky, token-major out
__global__ __launch_bounds__(256) void split_rms_t_kernel(
    const float* __restrict__ h1,    // [B, 3I, S]
    float* __restrict__ g2t)         // [NT, I]
{
    __shared__ float red[8][33];
    __shared__ float rinv[32];
    __shared__ float tile[32][33];
    const int b  = blockIdx.x / (SS / 32);
    const int s0 = (blockIdx.x % (SS / 32)) * 32;
    const int tx = threadIdx.x & 31;
    const int cg = threadIdx.x >> 5;

    float ss = 0.f;
    for (int i = cg; i < II; i += 8) {
        const float a0 = h1[((size_t)b * O0 + i) * SS + s0 + tx];
        const float a1 = h1[((size_t)b * O0 + II + i) * SS + s0 + tx];
        const float a2 = h1[((size_t)b * O0 + 2 * II + i) * SS + s0 + tx];
        const float z  = a0 * a1 + a2;
        ss += z * z;
    }
    red[cg][tx] = ss; __syncthreads();
    if (threadIdx.x < 32) {
        float tot = 0.f;
#pragma unroll
        for (int g = 0; g < 8; ++g) tot += red[g][threadIdx.x];
        rinv[threadIdx.x] = rsqrtf(tot / (float)II + 1e-6f);
    }
    __syncthreads();

    for (int i0 = 0; i0 < II; i0 += 32) {
#pragma unroll
        for (int r = 0; r < 32; r += 8) {
            const int i = i0 + cg + r;
            const float a0 = h1[((size_t)b * O0 + i) * SS + s0 + tx];
            const float a1 = h1[((size_t)b * O0 + II + i) * SS + s0 + tx];
            const float a2 = h1[((size_t)b * O0 + 2 * II + i) * SS + s0 + tx];
            const float v  = (a0 * a1 + a2) * rinv[tx];
            tile[cg + r][tx] = v > 0.f ? v : ALPHA * v;
        }
        __syncthreads();
#pragma unroll
        for (int r = 0; r < 32; r += 8)
            g2t[((size_t)b * SS + s0 + cg + r) * II + i0 + tx] = tile[tx][cg + r];
        __syncthreads();
    }
}

// =====================================================================
// 6) aux loss: sum_e mean(gates_e)*mean(one_hot_e), both MoEs. Fixed order.
// =====================================================================
__global__ __launch_bounds__(256) void loss_kernel(
    const float* __restrict__ gates0, const int* __restrict__ cnt0,
    const float* __restrict__ gates2, const int* __restrict__ cnt2,
    float* __restrict__ lossOut)
{
    __shared__ float red[256];
    const int tid = threadIdx.x;
    float loss = 0.f;
    for (int pass = 0; pass < 2; ++pass) {
        const float* gp = pass ? gates2 : gates0;
        const int*   cp = pass ? cnt2   : cnt0;
        for (int e = 0; e < EE; ++e) {
            float p = 0.f;
            for (int t = tid; t < NT; t += 256) p += gp[e * NT + t];
            red[tid] = p; __syncthreads();
            for (int off = 128; off > 0; off >>= 1) {
                if (tid < off) red[tid] += red[tid + off];
                __syncthreads();
            }
            if (tid == 0)
                loss += (red[0] / (float)NT) * ((float)cp[e] / (float)NT);
            __syncthreads();
        }
    }
    if (tid == 0) *lossOut = loss;
}

// =====================================================================
extern "C" void kernel_launch(void* const* d_in, const int* in_sizes, int n_in,
                              void* d_out, int out_size, void* d_ws, size_t ws_size,
                              hipStream_t stream) {
    (void)in_sizes; (void)n_in; (void)out_size; (void)ws_size;
    const float* x   = (const float*)d_in[0];   // [B,F,S]
    const float* w0g = (const float*)d_in[1];   // [F,E]
    const float* w0  = (const float*)d_in[2];   // [E,F,3I]
    const float* w1  = (const float*)d_in[3];   // [K,I/G,3I]
    const float* w2g = (const float*)d_in[4];   // [I,E]
    const float* w2  = (const float*)d_in[5];   // [E,I,F]
    float* out = (float*)d_out;                 // [B,F,S] + loss scalar

    char* ws = (char*)d_ws;
    const size_t szXT  = (size_t)NT * FF * sizeof(float);       //  6.29 MB
    const size_t szH0  = (size_t)BB * O0 * SS * sizeof(float);  // 37.75 MB
    const size_t szBIS = (size_t)BB * II * SS * sizeof(float);  // 12.58 MB
    float* xt  = (float*)(ws);
    float* h0  = (float*)(ws + szXT);
    float* y   = (float*)(ws + szXT + szH0);
    float* g1t = (float*)(ws + szXT + szH0 + szBIS);
    float* h1  = h0;                  // conv out reuses h0
    float* g2t = y;                   // split_norm out reuses y
    float* gates0 = (float*)(ws + szXT + szH0 + 2 * szBIS);
    float* gates2 = gates0 + EE * NT;
    int*   list0  = (int*)(gates2 + EE * NT);
    int*   list2  = list0 + EE * NT;
    int*   cnt0   = list2 + EE * NT;
    int*   cnt2   = cnt0 + EE;

    hipMemsetAsync(cnt0, 0, 2 * EE * sizeof(int), stream);

    // activations -> token-major
    transpose_xt_kernel<<<BB * (FF / 32) * (SS / 32), 256, 0, stream>>>(x, xt);

    // --- MoE #1: route + grouped GEMM F -> 3I
    gate_route_kernel<<<NT / 8, 256, 0, stream>>>(xt, w0g, FF, gates0, cnt0, list0);
    moe_gemm_kernel<<<EE * (O0 / 128) * (NT / 64), 256, 0, stream>>>(
        w0, xt, list0, cnt0, h0, FF, O0, O0 / 128, NT / 64);

    // --- linear-attention cumsum + rmsnorm + leaky (token-major out)
    cumsum_div_kernel<<<(BB * II) / 8, 256, 0, stream>>>(h0, y);
    rms_leaky_t_kernel<<<BB * (SS / 32), 256, 0, stream>>>(y, g1t);

    // --- causal grouped conv I -> 3I, then split_norm (token-major out)
    conv_gemm_kernel<<<BB * GG * 5 * (SS / 64), 256, 0, stream>>>(w1, g1t, h1);
    split_rms_t_kernel<<<BB * (SS / 32), 256, 0, stream>>>(h1, g2t);

    // --- MoE #2: route + grouped GEMM I -> F, straight into d_out
    gate_route_kernel<<<NT / 8, 256, 0, stream>>>(g2t, w2g, II, gates2, cnt2, list2);
    moe_gemm_kernel<<<EE * (FF / 128) * (NT / 64), 256, 0, stream>>>(
        w2, g2t, list2, cnt2, out, II, FF, FF / 128, NT / 64);

    // --- aux loss scalar appended after the [B,F,S] output
    loss_kernel<<<1, 256, 0, stream>>>(gates0, cnt0, gates2, cnt2,
                                       out + (size_t)BB * FF * SS);
}
